// GroupedMLP_30279519436989
// MI455X (gfx1250) — compile-verified, matrix operand order checked by reading, not run on hardware
//
#include <hip/hip_runtime.h>
#include <hip/hip_bf16.h>
#include <stdint.h>
#include <stddef.h>

// ---------------------------------------------------------------------------
// Grouped MoE MLP with SwiGLU for gfx1250 (MI455X), bf16 WMMA path.
//   o1 = x @ Wg ; o2 = x @ Wu ; inter = silu(o1)*o2 ; out = inter @ Wd
// Ragged per-expert row groups, tokens pre-sorted by expert.
// ---------------------------------------------------------------------------

typedef __bf16 bf16_t;
typedef __bf16 v16bf __attribute__((ext_vector_type(16)));
typedef __bf16 v8bf  __attribute__((ext_vector_type(8)));
typedef float  v8f   __attribute__((ext_vector_type(8)));

#define NE 8
#define NT 8192
#define NH 2048
#define NI 4096

#define BM 128          // rows per block tile
#define BN 64           // cols per block tile
#define BK 32           // K per step (== WMMA bf16 K)
#define BP32 20         // transposed-B LDS pitch in dwords (80B, keeps 16B align)

// Async global->LDS copy support (gfx1250): guard on builtin availability.
#if defined(__has_builtin)
#  if __has_builtin(__builtin_amdgcn_global_load_async_to_lds_b128)
#    define HAVE_ASYNC_COPY 1
#  endif
#endif
#ifndef HAVE_ASYNC_COPY
#  define HAVE_ASYNC_COPY 0
#endif

// Exact pointee type from hipcc diagnostic: int __attribute__((vector_size(16)))
typedef int v4i_vs __attribute__((vector_size(16)));
typedef __attribute__((address_space(1))) v4i_vs as1_v4i;  // global
typedef __attribute__((address_space(3))) v4i_vs as3_v4i;  // LDS

__device__ __forceinline__ void wait_asynccnt0() {
#if defined(__has_builtin) && __has_builtin(__builtin_amdgcn_s_wait_asynccnt)
  __builtin_amdgcn_s_wait_asynccnt(0);
#else
  asm volatile("s_wait_asynccnt 0x0" ::: "memory");
#endif
}

// fp32 -> bf16 round-to-nearest-even, raw 16 bits
__device__ __forceinline__ uint32_t f2bf_bits(float f) {
  uint32_t u = __float_as_uint(f);
  return (u + 0x7FFFu + ((u >> 16) & 1u)) >> 16;
}
__device__ __forceinline__ uint32_t pack2bf(float lo, float hi) {
  return f2bf_bits(lo) | (f2bf_bits(hi) << 16);
}
__device__ __forceinline__ bf16_t f2bf(float f) {
  union { unsigned short s; bf16_t b; } cv;
  cv.s = (unsigned short)f2bf_bits(f);
  return cv.b;
}

// Assemble the 16x32 bf16 WMMA A/B fragment from two contiguous 16B LDS runs.
// Per ISA: lane half 0 -> K 0..7 (lo) / 16..23 (hi); half 1 -> 8..15 / 24..31.
__device__ __forceinline__ v16bf make_frag(const void* lo, const void* hi) {
  union { v16bf v; v8bf h[2]; } u;
  u.h[0] = *(const v8bf*)lo;
  u.h[1] = *(const v8bf*)hi;
  return u.v;
}

__device__ __forceinline__ v8f wmma_bf16(v16bf a, v16bf b, v8f c) {
  return __builtin_amdgcn_wmma_f32_16x16x32_bf16(
      false, a, false, b, (short)0, c, false, false);
}

// ---------------------------------------------------------------------------
// Kernel 1: gate & up GEMMs fused with SwiGLU; writes bf16 inter [T, I].
// grid: (NI/BN, NT/BM, NE), block: 256 (8 waves)
// ---------------------------------------------------------------------------
__global__ __launch_bounds__(256) void moe_gateup_swiglu(
    const float* __restrict__ x,     // [T, H]
    const int*   __restrict__ tpe,   // [E]
    const float* __restrict__ Wg,    // [E, H, I]
    const float* __restrict__ Wu,    // [E, H, I]
    bf16_t*      __restrict__ inter) // [T, I] (workspace)
{
  const int e  = blockIdx.z;
  const int rt = blockIdx.y;
  const int ct = blockIdx.x;

  int start = 0;
  for (int i = 0; i < e; ++i) start += tpe[i];
  const int rowsE = tpe[e];
  if (rt * BM >= rowsE) return;

  const int n0 = ct * BN;

  __shared__ __align__(16) bf16_t   Alds[BM][BK];   // row-major [m][k]
  __shared__ __align__(16) uint32_t Bg32[BN][BP32]; // transposed [n][k/2] (k-pairs)
  __shared__ __align__(16) uint32_t Bu32[BN][BP32];

  const int tid  = threadIdx.x;
  const int lane = tid & 31;
  const int wid  = tid >> 5;
  const int wm   = wid & 3;    // 4 row-groups of 32
  const int wn   = wid >> 2;   // 2 col-groups of 32
  const int half = lane >> 4;
  const int l16  = lane & 15;

  v8f accG[2][2], accU[2][2];
  #pragma unroll
  for (int i = 0; i < 2; ++i)
    #pragma unroll
    for (int j = 0; j < 2; ++j)
      #pragma unroll
      for (int v = 0; v < 8; ++v) { accG[i][j][v] = 0.f; accU[i][j][v] = 0.f; }

  const float* WgE = Wg + (size_t)e * NH * NI;
  const float* WuE = Wu + (size_t)e * NH * NI;
  const int rowsClamp = rowsE - 1;   // rowsE >= 1 here

  // B staging decomposition: 256 threads x (4n x 2k) blocks covers 32x64 tile.
  const int nb4 = (tid & 15) * 4;    // n offset
  const int kb  = tid >> 4;          // k-pair index, 0..15
  const int k2  = kb * 2;            // k offset (even)

  for (int k0 = 0; k0 < NH; k0 += BK) {
    // ---- stage A: 128x32 fp32 -> bf16, float4 loads, branchless clamp ----
    #pragma unroll
    for (int i = 0; i < 4; ++i) {
      const int u  = tid + i * 256;      // unit = float4
      const int r  = u >> 3;             // 8 float4 per 32-float row
      const int c4 = (u & 7) * 4;
      const int rl = rt * BM + r;        // row within expert
      const int sr = (rl < rowsE) ? rl : rowsClamp;
      const float4 v = *(const float4*)&x[(size_t)(start + sr) * NH + k0 + c4];
      const bool ok = rl < rowsE;
      const float vx = ok ? v.x : 0.f, vy = ok ? v.y : 0.f;
      const float vz = ok ? v.z : 0.f, vw = ok ? v.w : 0.f;
      uint2 p;
      p.x = pack2bf(vx, vy);
      p.y = pack2bf(vz, vw);
      *(uint2*)&Alds[r][c4] = p;         // one ds_store_b64
    }
    // ---- stage B (gate & up): two k-rows per thread, pack k-pairs ----
    {
      const size_t gi0 = (size_t)(k0 + k2) * NI + n0 + nb4;
      const float4 g0 = *(const float4*)&WgE[gi0];
      const float4 g1 = *(const float4*)&WgE[gi0 + NI];
      const float4 u0 = *(const float4*)&WuE[gi0];
      const float4 u1 = *(const float4*)&WuE[gi0 + NI];
      if (k0 + BK < NH) {                // prefetch next K tile (global_prefetch_b8)
        __builtin_prefetch(&WgE[gi0 + (size_t)BK * NI], 0, 0);
        __builtin_prefetch(&WuE[gi0 + (size_t)BK * NI], 0, 0);
      }
      Bg32[nb4 + 0][kb] = pack2bf(g0.x, g1.x);
      Bg32[nb4 + 1][kb] = pack2bf(g0.y, g1.y);
      Bg32[nb4 + 2][kb] = pack2bf(g0.z, g1.z);
      Bg32[nb4 + 3][kb] = pack2bf(g0.w, g1.w);
      Bu32[nb4 + 0][kb] = pack2bf(u0.x, u1.x);
      Bu32[nb4 + 1][kb] = pack2bf(u0.y, u1.y);
      Bu32[nb4 + 2][kb] = pack2bf(u0.z, u1.z);
      Bu32[nb4 + 3][kb] = pack2bf(u0.w, u1.w);
    }
    __syncthreads();

    #pragma unroll
    for (int tm = 0; tm < 2; ++tm) {
      const int ar = wm * 32 + tm * 16 + l16;
      const v16bf af = make_frag(&Alds[ar][half * 8], &Alds[ar][16 + half * 8]);
      #pragma unroll
      for (int tn = 0; tn < 2; ++tn) {
        const int bc = wn * 32 + tn * 16 + l16;
        const v16bf bg = make_frag(&Bg32[bc][half * 4], &Bg32[bc][8 + half * 4]);
        const v16bf bu = make_frag(&Bu32[bc][half * 4], &Bu32[bc][8 + half * 4]);
        accG[tm][tn] = wmma_bf16(af, bg, accG[tm][tn]);
        accU[tm][tn] = wmma_bf16(af, bu, accU[tm][tn]);
      }
    }
    __syncthreads();
  }

  // ---- SwiGLU on accumulators, store bf16 to inter[T, I] ----
  #pragma unroll
  for (int tm = 0; tm < 2; ++tm) {
    #pragma unroll
    for (int tn = 0; tn < 2; ++tn) {
      const int col = n0 + wn * 32 + tn * 16 + l16;
      #pragma unroll
      for (int v = 0; v < 8; ++v) {
        const int rloc = wm * 32 + tm * 16 + v + half * 8;
        if (rt * BM + rloc < rowsE) {
          const float g = accG[tm][tn][v];
          const float u = accU[tm][tn][v];
          const float s = g / (1.f + __expf(-g));  // silu(g)
          inter[(size_t)(start + rt * BM + rloc) * NI + col] = f2bf(s * u);
        }
      }
    }
  }
}

// ---------------------------------------------------------------------------
// Kernel 2: down projection: out[T, H] = inter[T, I] @ Wd[e][I, H], fp32 out.
// grid: (NH/BN, NT/BM, NE), block: 256
// ---------------------------------------------------------------------------
__global__ __launch_bounds__(256) void moe_down(
    const bf16_t* __restrict__ inter, // [T, I] bf16
    const int*    __restrict__ tpe,   // [E]
    const float*  __restrict__ Wd,    // [E, I, H]
    float*        __restrict__ out)   // [T, H]
{
  const int e  = blockIdx.z;
  const int rt = blockIdx.y;
  const int ct = blockIdx.x;

  int start = 0;
  for (int i = 0; i < e; ++i) start += tpe[i];
  const int rowsE = tpe[e];
  if (rt * BM >= rowsE) return;

  const int n0 = ct * BN;

  __shared__ __align__(16) bf16_t   Alds[BM][BK];
  __shared__ __align__(16) uint32_t Bt32[BN][BP32];

  const int tid  = threadIdx.x;
  const int lane = tid & 31;
  const int wid  = tid >> 5;
  const int wm   = wid & 3;
  const int wn   = wid >> 2;
  const int half = lane >> 4;
  const int l16  = lane & 15;

  v8f acc[2][2];
  #pragma unroll
  for (int i = 0; i < 2; ++i)
    #pragma unroll
    for (int j = 0; j < 2; ++j)
      #pragma unroll
      for (int v = 0; v < 8; ++v) acc[i][j][v] = 0.f;

  const float* WdE = Wd + (size_t)e * NI * NH;
  const int rowsClamp = rowsE - 1;

  const int nb4 = (tid & 15) * 4;
  const int kb  = tid >> 4;
  const int k2  = kb * 2;

  for (int k0 = 0; k0 < NI; k0 += BK) {
    // ---- stage A: 128x32 bf16, pure copy global -> LDS (async DMA) ----
    #pragma unroll
    for (int i = 0; i < 2; ++i) {
      const int u  = tid + i * 256;     // unit = 16B (8 bf16)
      const int r  = u >> 2;            // 4 units per 32-elem row
      const int c  = (u & 3) * 8;
      const int rl = rt * BM + r;
#if HAVE_ASYNC_COPY
      if (rl < rowsE) {
        const bf16_t* gp = &inter[(size_t)(start + rl) * NI + k0 + c];
        bf16_t* lp = &Alds[r][c];
        __builtin_amdgcn_global_load_async_to_lds_b128(
            (as1_v4i*)(uintptr_t)gp,
            (as3_v4i*)(uint32_t)(uintptr_t)lp, 0, 0);
      } else {
        uint4 z; z.x = 0; z.y = 0; z.z = 0; z.w = 0;
        *(uint4*)&Alds[r][c] = z;
      }
#else
      const int sr = (rl < rowsE) ? rl : rowsClamp;
      union { v8bf v; uint4 q; } d;
      d.v = *(const v8bf*)&inter[(size_t)(start + sr) * NI + k0 + c];
      if (rl >= rowsE) { d.q.x = 0; d.q.y = 0; d.q.z = 0; d.q.w = 0; }
      *(v8bf*)&Alds[r][c] = d.v;
#endif
    }
    // ---- stage B: two k-rows per thread, pack k-pairs ----
    {
      const size_t gi0 = (size_t)(k0 + k2) * NH + n0 + nb4;
      const float4 w0 = *(const float4*)&WdE[gi0];
      const float4 w1 = *(const float4*)&WdE[gi0 + NH];
      if (k0 + BK < NI)
        __builtin_prefetch(&WdE[gi0 + (size_t)BK * NH], 0, 0);
      Bt32[nb4 + 0][kb] = pack2bf(w0.x, w1.x);
      Bt32[nb4 + 1][kb] = pack2bf(w0.y, w1.y);
      Bt32[nb4 + 2][kb] = pack2bf(w0.z, w1.z);
      Bt32[nb4 + 3][kb] = pack2bf(w0.w, w1.w);
    }
#if HAVE_ASYNC_COPY
    wait_asynccnt0();
#endif
    __syncthreads();

    #pragma unroll
    for (int tm = 0; tm < 2; ++tm) {
      const int ar = wm * 32 + tm * 16 + l16;
      const v16bf af = make_frag(&Alds[ar][half * 8], &Alds[ar][16 + half * 8]);
      #pragma unroll
      for (int tn = 0; tn < 2; ++tn) {
        const int bc = wn * 32 + tn * 16 + l16;
        const v16bf bf = make_frag(&Bt32[bc][half * 4], &Bt32[bc][8 + half * 4]);
        acc[tm][tn] = wmma_bf16(af, bf, acc[tm][tn]);
      }
    }
    __syncthreads();
  }

  // ---- store fp32 result ----
  #pragma unroll
  for (int tm = 0; tm < 2; ++tm) {
    #pragma unroll
    for (int tn = 0; tn < 2; ++tn) {
      const int col = n0 + wn * 32 + tn * 16 + l16;
      #pragma unroll
      for (int v = 0; v < 8; ++v) {
        const int rloc = wm * 32 + tm * 16 + v + half * 8;
        if (rt * BM + rloc < rowsE)
          out[(size_t)(start + rt * BM + rloc) * NH + col] = acc[tm][tn][v];
      }
    }
  }
}

// ---------------------------------------------------------------------------
extern "C" void kernel_launch(void* const* d_in, const int* in_sizes, int n_in,
                              void* d_out, int out_size, void* d_ws, size_t ws_size,
                              hipStream_t stream) {
  const float* x   = (const float*)d_in[0];  // [T, H]
  const int*   tpe = (const int*)  d_in[1];  // [E]
  const float* wg  = (const float*)d_in[2];  // [E, H, I]
  const float* wu  = (const float*)d_in[3];  // [E, H, I]
  const float* wd  = (const float*)d_in[4];  // [E, I, H]
  float* out = (float*)d_out;                // [T, H]

  bf16_t* inter = (bf16_t*)d_ws;             // [T, I] bf16 = 64 MB scratch
  (void)in_sizes; (void)n_in; (void)out_size; (void)ws_size;

  dim3 blk(256, 1, 1);
  dim3 g1(NI / BN, NT / BM, NE);
  moe_gateup_swiglu<<<g1, blk, 0, stream>>>(x, tpe, wg, wu, inter);

  dim3 g2(NH / BN, NT / BM, NE);
  moe_down<<<g2, blk, 0, stream>>>(inter, tpe, wd, out);
}
